// GraphVAE_67826123538494
// MI455X (gfx1250) — matrix operand order repaired and checked by reading, not compiled, not run on hardware
//
#include <hip/hip_runtime.h>

typedef _Float16 v16h __attribute__((ext_vector_type(16)));
typedef _Float16 v8h  __attribute__((ext_vector_type(8)));
typedef float    v8f  __attribute__((ext_vector_type(8)));

#define HID 128

// ---------------------------------------------------------------- utilities
__global__ void zero_f32(float* __restrict__ p, int n) {
    int i = blockIdx.x * blockDim.x + threadIdx.x;
    if (i < n) p[i] = 0.0f;
}

// deg[dst] += 1 over all E edges (deg buffer pre-zeroed; self-loop added later)
__global__ void deg_accum(const int* __restrict__ dst, float* __restrict__ deg, int E) {
    int e = blockIdx.x * blockDim.x + threadIdx.x;
    if (e < E) atomicAdd(&deg[dst[e]], 1.0f);
}

// dis[i] = rsqrt(deg[i] + 1)  (the +1 is the self loop; deg >= 1 always)
__global__ void deg_to_dis(float* __restrict__ d, int N) {
    int i = blockIdx.x * blockDim.x + threadIdx.x;
    if (i < N) d[i] = rsqrtf(d[i] + 1.0f);
}

// ------------------------------------------------- WMMA GEMM: C = A[N,K] @ W[K,128]
// Block = 256 threads (8 waves): 16 output rows x 128 cols.
// A tile and W staged in LDS as f16 (W transposed so fragments are contiguous);
// each wave computes one 16x16 tile with K/32 fully-unrolled WMMAs.
template <int K>
__global__ void gemm_xw_wmma(const float* __restrict__ A, const float* __restrict__ W,
                             float* __restrict__ C, int Nrows) {
    __shared__ _Float16 Ah[16][K + 8];    // +8 halves row pad (bank skew, keeps 16B align)
    __shared__ _Float16 Wt[HID][K + 8];   // W transposed: Wt[col][k]
    const int t = threadIdx.x;
    const int row0 = blockIdx.x << 4;

    // stage A tile (16 x K), f32 -> f16
    for (int idx = t; idx < 16 * K; idx += 256) {
        int r = idx / K, k = idx - r * K;
        int gr = row0 + r; if (gr >= Nrows) gr = Nrows - 1;
        Ah[r][k] = (_Float16)A[(size_t)gr * K + k];
    }
    // stage W (K x 128) transposed, f32 -> f16 (coalesced global reads)
    for (int idx = t; idx < K * HID; idx += 256) {
        int k = idx >> 7, c = idx & (HID - 1);
        Wt[c][k] = (_Float16)W[(size_t)k * HID + c];
    }
    __syncthreads();

    const int lane = t & 31, wave = t >> 5;
    const int col0 = wave << 4;
    const int m = lane & 15;              // A row / B,C col within tile
    const int g = lane >> 4;              // lane group
    v8f acc = {};
#pragma unroll
    for (int k0 = 0; k0 < K; k0 += 32) {
        // A fragment: halves [k0+8g .. +7] then [k0+16+8g .. +7]  (two b128 LDS loads)
        v8h alo = *(const v8h*)&Ah[m][k0 + 8 * g];
        v8h ahi = *(const v8h*)&Ah[m][k0 + 16 + 8 * g];
        // B fragment: lane holds column col0+m, K = [k0+8g .. +7] then [k0+16+8g .. +7]
        v8h blo = *(const v8h*)&Wt[col0 + m][k0 + 8 * g];
        v8h bhi = *(const v8h*)&Wt[col0 + m][k0 + 16 + 8 * g];
        v16h a = __builtin_shufflevector(alo, ahi, 0, 1, 2, 3, 4, 5, 6, 7,
                                         8, 9, 10, 11, 12, 13, 14, 15);
        v16h b = __builtin_shufflevector(blo, bhi, 0, 1, 2, 3, 4, 5, 6, 7,
                                         8, 9, 10, 11, 12, 13, 14, 15);
        acc = __builtin_amdgcn_wmma_f32_16x16x32_f16(false, a, false, b,
                                                     (short)0, acc, false, false);
    }
    // C/D: VGPR j -> row j + 8*g, col m. Fast path: whole tile in range ->
    // 8 unguarded stores (immediate offsets, single clause).
    const int orow = row0 + (g << 3);
    float* cp = C + (size_t)orow * HID + col0 + m;
    if (orow + 8 <= Nrows) {
#pragma unroll
        for (int j = 0; j < 8; ++j) cp[j * HID] = acc[j];
    } else {
#pragma unroll
        for (int j = 0; j < 8; ++j)
            if (orow + j < Nrows) cp[j * HID] = acc[j];
    }
}

// ------------------------------------------- message passing: one wave per edge
__global__ void edge_scatter(const float* __restrict__ h, const int* __restrict__ src,
                             const int* __restrict__ dst, const float* __restrict__ dis,
                             float* __restrict__ agg, int E) {
    int wid  = (blockIdx.x * blockDim.x + threadIdx.x) >> 5;
    int lane = threadIdx.x & 31;
    if (wid >= E) return;
    int s = src[wid], d = dst[wid];
    float nrm = dis[s] * dis[d];
    const float4 v = ((const float4*)(h + (size_t)s * HID))[lane]; // 4 channels/lane
    float* ap = agg + (size_t)d * HID + (lane << 2);
    atomicAdd(ap + 0, v.x * nrm);
    atomicAdd(ap + 1, v.y * nrm);
    atomicAdd(ap + 2, v.z * nrm);
    atomicAdd(ap + 3, v.w * nrm);
}

// self-loop + bias + ReLU, in place on agg
__global__ void gcn_finalize(float* __restrict__ agg, const float* __restrict__ h,
                             const float* __restrict__ dis, const float* __restrict__ bias,
                             int total) {
    int idx = blockIdx.x * blockDim.x + threadIdx.x;
    if (idx >= total) return;
    int i = idx >> 7;
    int c = idx & (HID - 1);
    float dii = dis[i];
    float v = agg[idx] + h[idx] * dii * dii + bias[c];
    agg[idx] = fmaxf(v, 0.0f);
}

// per-channel column sum of [N,128] into gsum[128] (pre-zeroed)
__global__ void col_sum(const float* __restrict__ h, float* __restrict__ gsum, int N) {
    int c = threadIdx.x & (HID - 1);
    int r = (blockIdx.x << 1) + (threadIdx.x >> 7);
    int stride = gridDim.x << 1;
    float s = 0.0f;
    for (; r < N; r += stride) s += h[(size_t)r * HID + c];
    atomicAdd(&gsum[c], s);
}

// mean -> mu/logvar -> z -> decoder fc1 (all tiny; one block of 128 threads)
__global__ void vae_head(const float* __restrict__ gsum, const float* __restrict__ eps,
                         const float* __restrict__ Wmu, const float* __restrict__ bmu,
                         const float* __restrict__ Wlv, const float* __restrict__ blv,
                         const float* __restrict__ Wd1, const float* __restrict__ bd1,
                         float* __restrict__ mu_out, float* __restrict__ lv_out,
                         float* __restrict__ dvec, float invN) {
    __shared__ float gs[HID];
    __shared__ float zs[64];
    int t = threadIdx.x; // 128 threads
    gs[t] = gsum[t] * invN;
    __syncthreads();
    if (t < 64) {
        float mu = bmu[t], lv = blv[t];
#pragma unroll 4
        for (int h = 0; h < HID; ++h) {
            mu = fmaf(gs[h], Wmu[h * 64 + t], mu);
            lv = fmaf(gs[h], Wlv[h * 64 + t], lv);
        }
        mu_out[t] = mu;
        lv_out[t] = lv;
        zs[t] = mu + eps[t] * expf(0.5f * lv);
    }
    __syncthreads();
    float acc = bd1[t];
#pragma unroll 4
    for (int l = 0; l < 64; ++l) acc = fmaf(zs[l], Wd1[l * HID + t], acc);
    dvec[t] = fmaxf(acc, 0.0f);
}

// recon[n..n+3] = d[128] . Wd2[:,n..n+3] + bd2[n..n+3]
// Streams 819MB of Wd2 with b128 loads — the HBM roofline kernel (~35us floor).
__global__ void decode_fc2(const float* __restrict__ dvec, const float* __restrict__ Wd2,
                           const float* __restrict__ bd2, float* __restrict__ out, int NC) {
    __shared__ float ds_[HID];
    int t = threadIdx.x;
    if (t < HID) ds_[t] = dvec[t];
    __syncthreads();
    int n = (blockIdx.x * blockDim.x + t) << 2;
    if (n + 4 <= NC) {
        float4 acc = *(const float4*)(bd2 + n);
        const float* wp = Wd2 + n;
#pragma unroll 4
        for (int h = 0; h < HID; ++h) {
            if (h + 16 < HID) __builtin_prefetch(wp + (size_t)(h + 16) * NC, 0, 0);
            float4 w = *(const float4*)(wp + (size_t)h * NC);
            float dh = ds_[h];
            acc.x = fmaf(dh, w.x, acc.x);
            acc.y = fmaf(dh, w.y, acc.y);
            acc.z = fmaf(dh, w.z, acc.z);
            acc.w = fmaf(dh, w.w, acc.w);
        }
        *(float4*)(out + n) = acc;
    } else {
        for (int j = 0; n + j < NC; ++j) {
            float acc = bd2[n + j];
            for (int h = 0; h < HID; ++h)
                acc = fmaf(ds_[h], Wd2[(size_t)h * NC + n + j], acc);
            out[n + j] = acc;
        }
    }
}

// ---------------------------------------------------------------- launcher
extern "C" void kernel_launch(void* const* d_in, const int* in_sizes, int n_in,
                              void* d_out, int out_size, void* d_ws, size_t ws_size,
                              hipStream_t stream) {
    const float* x   = (const float*)d_in[0];
    const int*   ei  = (const int*)d_in[1];
    const float* eps = (const float*)d_in[2];
    const float* W1  = (const float*)d_in[3];
    const float* b1  = (const float*)d_in[4];
    const float* W2  = (const float*)d_in[5];
    const float* b2  = (const float*)d_in[6];
    const float* Wmu = (const float*)d_in[7];
    const float* bmu = (const float*)d_in[8];
    const float* Wlv = (const float*)d_in[9];
    const float* blv = (const float*)d_in[10];
    const float* Wd1 = (const float*)d_in[11];
    const float* bd1 = (const float*)d_in[12];
    const float* Wd2 = (const float*)d_in[13];
    const float* bd2 = (const float*)d_in[14];
    float* out = (float*)d_out;

    const int CIN = 32, LAT = 64;
    const int Nn = in_sizes[0] / CIN;     // 50000
    const int E  = in_sizes[1] / 2;       // 1600000
    const int NC = in_sizes[14];          // N*C_IN = 1600000
    const int* srcI = ei;
    const int* dstI = ei + E;
    const int NH = Nn * HID;              // 6.4M

    float* wsf  = (float*)d_ws;
    float* dis  = wsf;
    float* bufA = wsf + Nn;               // h buffer  [N,128]
    float* bufB = bufA + (size_t)NH;      // agg / activations [N,128]
    float* gsum = bufB + (size_t)NH;      // [128]
    float* dvec = gsum + HID;             // [128]

    const int T = 256;
    // normalization
    zero_f32<<<(Nn + T - 1) / T, T, 0, stream>>>(dis, Nn);
    deg_accum<<<(E + T - 1) / T, T, 0, stream>>>(dstI, dis, E);
    deg_to_dis<<<(Nn + T - 1) / T, T, 0, stream>>>(dis, Nn);

    // GCN layer 1: h1 = x @ W1 ; agg ; self-loop+bias+relu
    gemm_xw_wmma<32><<<(Nn + 15) / 16, T, 0, stream>>>(x, W1, bufA, Nn);
    zero_f32<<<(NH + T - 1) / T, T, 0, stream>>>(bufB, NH);
    edge_scatter<<<(E + 7) / 8, T, 0, stream>>>(bufA, srcI, dstI, dis, bufB, E);
    gcn_finalize<<<(NH + T - 1) / T, T, 0, stream>>>(bufB, bufA, dis, b1, NH);

    // GCN layer 2: h2 = relu1 @ W2 ; agg ; self-loop+bias+relu
    gemm_xw_wmma<128><<<(Nn + 15) / 16, T, 0, stream>>>(bufB, W2, bufA, Nn);
    zero_f32<<<(NH + T - 1) / T, T, 0, stream>>>(bufB, NH);
    edge_scatter<<<(E + 7) / 8, T, 0, stream>>>(bufA, srcI, dstI, dis, bufB, E);
    gcn_finalize<<<(NH + T - 1) / T, T, 0, stream>>>(bufB, bufA, dis, b2, NH);

    // graph readout + VAE head
    zero_f32<<<1, T, 0, stream>>>(gsum, HID);
    col_sum<<<256, T, 0, stream>>>(bufB, gsum, Nn);
    vae_head<<<1, HID, 0, stream>>>(gsum, eps, Wmu, bmu, Wlv, blv, Wd1, bd1,
                                    out + NC, out + NC + LAT, dvec, 1.0f / (float)Nn);

    // decoder fc2 (HBM-bound streaming GEMV, float4 per thread)
    decode_fc2<<<(NC / 4 + T - 1) / T, T, 0, stream>>>(dvec, Wd2, bd2, out, NC);
}
